// Network_74835510165675
// MI455X (gfx1250) — matrix operand order repaired and checked by reading, not compiled, not run on hardware
//
#include <hip/hip_runtime.h>
#include <cstdint>

// ---------------------------------------------------------------------------
// 2-layer tanh RNN (B=20, T=2048, IN=256, H=512) + final Linear(512->128)
// CDNA5 / gfx1250: bf16 WMMA (v_wmma_f32_16x16x32_bf16) everywhere, persistent
// cluster-style recurrence kernel with LDS-resident Whh slices and TDM
// (tensor_load_to_lds) staging of the recurrent state each timestep.
// ---------------------------------------------------------------------------

#define IN_SZ 256
#define HID   512
#define OUTSZ 128
#define BATCH 20
#define BPAD  32      // batch padded to 2 M-tiles of 16
#define SEQ   2048
#define NWG   16      // cooperating workgroups in the recurrence
#define RNN_THREADS 128
#define USE_TDM 1     // 1: stage h via tensor_load_to_lds (TDM); 0: scalar loads

typedef __attribute__((ext_vector_type(16))) __bf16 v16bf;
typedef __attribute__((ext_vector_type(8)))  float  v8f;
typedef __attribute__((ext_vector_type(4)))  unsigned v4u;
typedef __attribute__((ext_vector_type(8)))  unsigned v8u;

union Frag16 {
  uint4              q[2];   // 32 bytes = 16 bf16 per lane
  unsigned long long d[4];
  v16bf              v;
};

__device__ __forceinline__ uint16_t f2bf(float f) {
  uint32_t u = __float_as_uint(f);
  u += 0x7FFFu + ((u >> 16) & 1u);          // round-to-nearest-even
  return (uint16_t)(u >> 16);
}
__device__ __forceinline__ uint32_t pk2(float a, float b) {
  return (uint32_t)f2bf(a) | ((uint32_t)f2bf(b) << 16);
}

__device__ __forceinline__ v8f wmma_bf16(const Frag16& A, const Frag16& B, v8f c) {
  // D = A(16x32 bf16) * B(32x16 bf16) + C(16x16 f32)
  return __builtin_amdgcn_wmma_f32_16x16x32_bf16(false, A.v, false, B.v,
                                                 (short)0, c, false, false);
}

#if USE_TDM
// ---------------------------------------------------------------------------
// TDM: one-shot DMA of a rows x cols bf16 tile (tile == tensor, row stride =
// cols elements) from global memory into LDS.  D# per CDNA5 ISA section 8:
//   group0: count=1 | lds_addr | global_addr[56:0] | type=2
//   group1: data_size=1(2B) | tensor_dim0=cols | tensor_dim1=rows |
//           tile_dim0=cols | tile_dim1=rows | tensor_dim0_stride=cols
// 2D tensor -> VADDR2/VADDR3 omitted (NULL).  Tracked by TENSORcnt.
// ---------------------------------------------------------------------------
__device__ __forceinline__ void tdm_load_2d_to_lds(unsigned lds_addr,
                                                   const void* gaddr,
                                                   unsigned rows, unsigned cols) {
  unsigned long long ga = (unsigned long long)gaddr;
  v4u g0;
  g0.x = 0x1u;                                  // count=1, user descriptor
  g0.y = lds_addr;                              // LDS byte address
  g0.z = (unsigned)ga;                          // global_addr[31:0]
  g0.w = ((unsigned)(ga >> 32) & 0x01FFFFFFu)   // global_addr[56:32]
         | (2u << 30);                          // type = 2 ("image")
  v8u g1;
  g1[0] = (1u << 16);                           // wg_mask=0, data_size=1 (2B)
  g1[1] = (cols & 0xFFFFu) << 16;               // tensor_dim0[15:0]
  g1[2] = ((cols >> 16) & 0xFFFFu)              // tensor_dim0[31:16]
          | ((rows & 0xFFFFu) << 16);           // tensor_dim1[15:0]
  g1[3] = ((rows >> 16) & 0xFFFFu)              // tensor_dim1[31:16]
          | ((cols & 0xFFFFu) << 16);           // tile_dim0
  g1[4] = rows & 0xFFFFu;                       // tile_dim1 (tile_dim2=0)
  g1[5] = cols;                                 // tensor_dim0_stride[31:0]
  g1[6] = 0u;                                   // stride hi, dim1_stride lo
  g1[7] = 0u;
  asm volatile("tensor_load_to_lds %0, %1" :: "s"(g0), "s"(g1) : "memory");
}
#endif

// ---------------------------------------------------------------------------
// Convert an f32 weight matrix W[N][K] (row-major, C = A*W^T) into bf16
// B-fragment order:  frag tile (n_tile, k_tile) is 1024 bytes:
//   [half(2)][lane(32)][j8(8)] bf16, where element j = half*8+j8 of lane L is
//   B[k = k_tile*32 + 8*(L/16) + j + 8*half][n = n_tile*16 + L%16]  (B = W^T)
// so in-kernel each lane fetches its fragment as two coalesced 16B loads.
// ---------------------------------------------------------------------------
__global__ void k_cvt_frag(const float* __restrict__ W, uint16_t* __restrict__ dst,
                           int N, int K) {
  int e = blockIdx.x * blockDim.x + threadIdx.x;
  if (e >= N * K) return;
  int tile = e >> 9;
  int w    = e & 511;
  int half = w >> 8;
  int lane = (w >> 3) & 31;
  int j    = half * 8 + (w & 7);
  int ktiles = K >> 5;
  int n_tile = tile / ktiles;
  int k_tile = tile - n_tile * ktiles;
  int col = n_tile * 16 + (lane & 15);
  int row = k_tile * 32 + ((lane >> 4) << 3) + j + (half << 3);
  dst[e] = f2bf(W[(size_t)col * K + row]);
}

// ---------------------------------------------------------------------------
// pre0[t][b][n] = sum_k x[b][t][k] * Wih0[n][k] + b_ih0[n]      (f32 out)
// grid = SEQ*16 blocks of 128 threads (4 waves); wave -> (m half, n tile)
// ---------------------------------------------------------------------------
__global__ void __launch_bounds__(128)
k_pre0(const float* __restrict__ x, const uint16_t* __restrict__ wfrag,
       const float* __restrict__ bias, float* __restrict__ pre) {
  const int blk  = blockIdx.x;
  const int t    = blk >> 4;
  const int nq   = blk & 15;
  const int wave = threadIdx.x >> 5;
  const int lane = threadIdx.x & 31;
  const int lo = lane & 15, hi = lane >> 4;
  const int m      = wave & 1;
  const int n_tile = nq * 2 + (wave >> 1);
  const int ncol   = n_tile * 16 + lo;
  const int b      = m * 16 + lo;

  Frag16 A[8];
  if (b < BATCH) {
    const float* row = x + ((size_t)b * SEQ + t) * IN_SZ;
#pragma unroll
    for (int kt = 0; kt < 8; ++kt) {
      const float* s = row + kt * 32 + hi * 8;      // K = kt*32 + 8*hi + {0..7,16..23}
      float4 x0 = *(const float4*)(s);
      float4 x1 = *(const float4*)(s + 4);
      float4 x2 = *(const float4*)(s + 16);
      float4 x3 = *(const float4*)(s + 20);
      A[kt].q[0] = make_uint4(pk2(x0.x,x0.y), pk2(x0.z,x0.w), pk2(x1.x,x1.y), pk2(x1.z,x1.w));
      A[kt].q[1] = make_uint4(pk2(x2.x,x2.y), pk2(x2.z,x2.w), pk2(x3.x,x3.y), pk2(x3.z,x3.w));
    }
  } else {
#pragma unroll
    for (int kt = 0; kt < 8; ++kt) {
      A[kt].q[0] = make_uint4(0u,0u,0u,0u);
      A[kt].q[1] = make_uint4(0u,0u,0u,0u);
    }
  }

  const float bv = bias[ncol];
  v8f acc;
#pragma unroll
  for (int r = 0; r < 8; ++r) acc[r] = bv;

  const uint4* fb = (const uint4*)wfrag;
#pragma unroll
  for (int kt = 0; kt < 8; ++kt) {
    Frag16 B;
    const uint4* bt = fb + (size_t)(n_tile * 8 + kt) * 64;
    B.q[0] = bt[lane];
    B.q[1] = bt[32 + lane];
    acc = wmma_bf16(A[kt], B, acc);
  }

  float* dst = pre + (size_t)t * BPAD * HID + ncol;
#pragma unroll
  for (int r = 0; r < 8; ++r) dst[(size_t)(r + hi * 8) * HID] = acc[r];
}

// ---------------------------------------------------------------------------
// pre1[t][b][n] = sum_k out0[t][b][k] * Wih1[n][k] + b_ih1[n]
// A comes from bf16 out0[t][32][512] -> A fragment = two contiguous b128 loads
// ---------------------------------------------------------------------------
__global__ void __launch_bounds__(128)
k_pre1(const uint16_t* __restrict__ a0, const uint16_t* __restrict__ wfrag,
       const float* __restrict__ bias, float* __restrict__ pre) {
  const int blk  = blockIdx.x;
  const int t    = blk >> 4;
  const int nq   = blk & 15;
  const int wave = threadIdx.x >> 5;
  const int lane = threadIdx.x & 31;
  const int lo = lane & 15, hi = lane >> 4;
  const int m      = wave & 1;
  const int n_tile = nq * 2 + (wave >> 1);
  const int ncol   = n_tile * 16 + lo;

  const uint4* arow = (const uint4*)(a0 + ((size_t)t * BPAD + m * 16 + lo) * HID);
  const float bv = bias[ncol];
  v8f acc0, acc1;
#pragma unroll
  for (int r = 0; r < 8; ++r) { acc0[r] = bv; acc1[r] = 0.0f; }

  const uint4* fb = (const uint4*)wfrag;
#pragma unroll
  for (int kt = 0; kt < 16; ++kt) {
    Frag16 A, B;
    int k0 = kt * 32 + hi * 8;                 // element index in the row
    A.q[0] = arow[k0 >> 3];                    // elements j0..7
    A.q[1] = arow[(k0 >> 3) + 2];              // elements j8..15 (K+16)
    const uint4* bt = fb + (size_t)(n_tile * 16 + kt) * 64;
    B.q[0] = bt[lane];
    B.q[1] = bt[32 + lane];
    if (kt & 1) acc1 = wmma_bf16(A, B, acc1);
    else        acc0 = wmma_bf16(A, B, acc0);
  }

  float* dst = pre + (size_t)t * BPAD * HID + ncol;
#pragma unroll
  for (int r = 0; r < 8; ++r) dst[(size_t)(r + hi * 8) * HID] = acc0[r] + acc1[r];
}

// ---------------------------------------------------------------------------
// Chip-wide barrier for the NWG persistent workgroups.  s_cluster_barrier is a
// NOP when not launched as a cluster (ISA 6.6); the atomic sense barrier is the
// guaranteed-correct path.
// ---------------------------------------------------------------------------
__device__ __forceinline__ void grid_barrier(int* cnt, int& phase) {
  __syncthreads();
  if (threadIdx.x < 32) {
    __builtin_amdgcn_s_cluster_barrier();     // s_barrier_signal/wait -3
  }
  ++phase;
  if (threadIdx.x == 0) {
    __hip_atomic_fetch_add(cnt, 1, __ATOMIC_ACQ_REL, __HIP_MEMORY_SCOPE_AGENT);
    while (__hip_atomic_load(cnt, __ATOMIC_ACQUIRE, __HIP_MEMORY_SCOPE_AGENT) < NWG * phase) {
      __builtin_amdgcn_s_sleep(1);
    }
  }
  __syncthreads();
}

// ---------------------------------------------------------------------------
// Persistent recurrence: h = tanh(pre[t] + h @ Whh^T + b_hh), t = 0..SEQ-1.
// NWG=16 workgroups of 128 threads; WG c owns output columns [c*32, c*32+32):
// its Whh fragment slice (32KB bf16) lives in LDS for the whole kernel.
// h (bf16 [32][512]) is exchanged through global hbuf each step; staging into
// LDS is one TDM tensor_load_to_lds per step (TENSORcnt-tracked).
// ---------------------------------------------------------------------------
__global__ void __launch_bounds__(RNN_THREADS)
k_rnn(const uint16_t* __restrict__ whh_frag,
      const float*    __restrict__ bias,       // b_hh
      const float*    __restrict__ pre,        // [SEQ][32][512] f32 (includes b_ih)
      uint16_t*       __restrict__ hbuf,       // [32][512] bf16 state
      uint16_t*       __restrict__ out_all,    // [SEQ][32][512] bf16 or null
      float*          __restrict__ hfinal,     // [32][512] f32 or null
      int*            __restrict__ barcnt) {
  __shared__ uint4              Wsl[2048];     // 32KB: 2 n-tiles x 16 k-tiles x 1KB
  __shared__ unsigned long long Hsl[4096];     // 32KB: full h as bf16

  const int tidx = threadIdx.x;
  const int c    = blockIdx.x;
  const int wave = tidx >> 5;
  const int lane = tidx & 31;
  const int lo = lane & 15, hi = lane >> 4;
  const int m      = wave & 1;                 // M half (rows 0-15 / 16-31)
  const int nl     = wave >> 1;                // local n-tile 0..1
  const int n_tile = c * 2 + nl;
  const int ncol   = n_tile * 16 + lo;
  const int Mrow   = m * 16 + lo;

  // Stage this WG's Whh fragment slice into LDS once.
  const uint4* fb = (const uint4*)whh_frag;
  for (int i = tidx; i < 2048; i += RNN_THREADS) {
    int ltile = i >> 6;                        // nloc*16 + kt
    int off   = i & 63;
    int nloc  = ltile >> 4;
    int kt    = ltile & 15;
    Wsl[i] = fb[(size_t)((c * 2 + nloc) * 16 + kt) * 64 + off];
  }
  const float bv = bias[ncol];
#if !USE_TDM
  const unsigned long long* hb64 = (const unsigned long long*)hbuf;
#endif
  int phase = 0;
  __syncthreads();

  for (int t = 0; t < SEQ; ++t) {
    // ---- stage h(t) into LDS ----
#if USE_TDM
    if (tidx == 0) {
      tdm_load_2d_to_lds((unsigned)(uintptr_t)Hsl, hbuf, BPAD, HID);
      __builtin_amdgcn_s_wait_tensorcnt(0);
    }
#else
    for (int i = tidx; i < 4096; i += RNN_THREADS) {
      Hsl[i] = __hip_atomic_load(hb64 + i, __ATOMIC_RELAXED, __HIP_MEMORY_SCOPE_AGENT);
    }
#endif
    __syncthreads();
    grid_barrier(barcnt, phase);               // everyone done reading hbuf

    v8f acc0, acc1;
#pragma unroll
    for (int r = 0; r < 8; ++r) { acc0[r] = bv; acc1[r] = 0.0f; }

    const float* prow = pre + (size_t)t * BPAD * HID + ncol;
    __builtin_prefetch(prow + (size_t)BPAD * HID, 0, 1);  // next step's pre row

#pragma unroll
    for (int kt = 0; kt < 16; ++kt) {
      Frag16 A, B;
      const int base = Mrow * 128 + kt * 8 + hi * 2;   // u64 index into Hsl
      A.d[0] = Hsl[base];
      A.d[1] = Hsl[base + 1];
      A.d[2] = Hsl[base + 4];                          // K + 16
      A.d[3] = Hsl[base + 5];
      const uint4* bt = &Wsl[(nl * 16 + kt) << 6];
      B.q[0] = bt[lane];
      B.q[1] = bt[32 + lane];
      if (kt & 1) acc1 = wmma_bf16(A, B, acc1);
      else        acc0 = wmma_bf16(A, B, acc0);
    }

    float fv[8];
#pragma unroll
    for (int r = 0; r < 8; ++r) {
      int M = r + hi * 8;
      fv[r] = tanhf(acc0[r] + acc1[r] + prow[(size_t)M * HID]);
    }

#pragma unroll
    for (int r = 0; r < 8; ++r) {
      int M = r + hi * 8;
      uint16_t hv = f2bf(fv[r]);
      hbuf[(size_t)M * HID + ncol] = hv;
      if (out_all) out_all[((size_t)t * BPAD + M) * HID + ncol] = hv;
      if (hfinal && t == SEQ - 1) hfinal[(size_t)M * HID + ncol] = fv[r];
    }
    __threadfence();                           // make h(t+1) device-visible
    grid_barrier(barcnt, phase);
  }
}

// ---------------------------------------------------------------------------
__global__ void k_init(uint32_t* __restrict__ hbuf32, int* __restrict__ barcnt) {
  int i = blockIdx.x * blockDim.x + threadIdx.x;
  if (i < BPAD * HID / 2) hbuf32[i] = 0u;
  if (i == 0) *barcnt = 0;
}

__global__ void k_fc(const float* __restrict__ h, const float* __restrict__ wfc,
                     const float* __restrict__ bfc, float* __restrict__ out) {
  int i = blockIdx.x * blockDim.x + threadIdx.x;
  if (i >= BATCH * OUTSZ) return;
  int b = i >> 7, o = i & 127;
  const float* hr = h + (size_t)b * HID;
  const float* wr = wfc + (size_t)o * HID;
  float s = bfc[o];
#pragma unroll 8
  for (int k = 0; k < HID; ++k) s = fmaf(hr[k], wr[k], s);
  out[i] = s;
}

// ---------------------------------------------------------------------------
extern "C" void kernel_launch(void* const* d_in, const int* in_sizes, int n_in,
                              void* d_out, int out_size, void* d_ws, size_t ws_size,
                              hipStream_t stream) {
  (void)in_sizes; (void)n_in; (void)out_size; (void)ws_size;
  const float* x    = (const float*)d_in[0];
  // d_in[1] = h0 (all zeros; handled by k_init)
  const float* Wih0 = (const float*)d_in[2];
  const float* Whh0 = (const float*)d_in[3];
  const float* bih0 = (const float*)d_in[4];
  const float* bhh0 = (const float*)d_in[5];
  const float* Wih1 = (const float*)d_in[6];
  const float* Whh1 = (const float*)d_in[7];
  const float* bih1 = (const float*)d_in[8];
  const float* bhh1 = (const float*)d_in[9];
  const float* Wfc  = (const float*)d_in[10];
  const float* bfc  = (const float*)d_in[11];

  char* ws = (char*)d_ws;
  size_t o = 0;
  auto take = [&](size_t bytes) -> char* {
    char* p = ws + o;
    o = (o + bytes + 255) & ~(size_t)255;
    return p;
  };
  uint16_t* fih0 = (uint16_t*)take((size_t)IN_SZ * HID * 2);   // 256 KB
  uint16_t* fhh0 = (uint16_t*)take((size_t)HID * HID * 2);     // 512 KB
  uint16_t* fih1 = (uint16_t*)take((size_t)HID * HID * 2);
  uint16_t* fhh1 = (uint16_t*)take((size_t)HID * HID * 2);
  uint16_t* hbuf = (uint16_t*)take((size_t)BPAD * HID * 2);    // 32 KB
  float*  hfinal = (float*)take((size_t)BPAD * HID * 4);       // 64 KB
  int*    barcnt = (int*)take(256);
  float*     pre = (float*)take((size_t)SEQ * BPAD * HID * 4); // 128 MB (reused for pre1)
  uint16_t* out0 = (uint16_t*)take((size_t)SEQ * BPAD * HID * 2); // 64 MB

  // Weight conversion into bf16 fragment layout.
  k_cvt_frag<<<(HID * IN_SZ + 255) / 256, 256, 0, stream>>>(Wih0, fih0, HID, IN_SZ);
  k_cvt_frag<<<(HID * HID + 255) / 256, 256, 0, stream>>>(Whh0, fhh0, HID, HID);
  k_cvt_frag<<<(HID * HID + 255) / 256, 256, 0, stream>>>(Wih1, fih1, HID, HID);
  k_cvt_frag<<<(HID * HID + 255) / 256, 256, 0, stream>>>(Whh1, fhh1, HID, HID);

  // Layer 0.
  k_pre0<<<SEQ * 16, 128, 0, stream>>>(x, fih0, bih0, pre);
  k_init<<<32, 256, 0, stream>>>((uint32_t*)hbuf, barcnt);
  k_rnn<<<NWG, RNN_THREADS, 0, stream>>>(fhh0, bhh0, pre, hbuf, out0, nullptr, barcnt);

  // Layer 1 (pre buffer reused).
  k_pre1<<<SEQ * 16, 128, 0, stream>>>(out0, fih1, bih1, pre);
  k_init<<<32, 256, 0, stream>>>((uint32_t*)hbuf, barcnt);
  k_rnn<<<NWG, RNN_THREADS, 0, stream>>>(fhh1, bhh1, pre, hbuf, nullptr, hfinal, barcnt);

  // Final Linear on last timestep.
  k_fc<<<(BATCH * OUTSZ + 255) / 256, 256, 0, stream>>>(hfinal, Wfc, bfc, (float*)d_out);
}
